// LSTMModel_14087492731648
// MI455X (gfx1250) — matrix operand order repaired and checked by reading, not compiled, not run on hardware
//
#include <hip/hip_runtime.h>
#include <hip/hip_bf16.h>

// ---------------------------------------------------------------------------
// Fused 2-layer LSTM (H=50, B=2048, T=512, in=1) + FC, persistent kernel.
// - One WG = 16 batch rows (one WMMA M tile), 4 waves split the 4 gates.
// - All weights (f16, padded to 256x64) + h/c state live in LDS (~131 KB).
// - Recurrent + input-projection GEMMs via v_wmma_f32_16x16x32_f16.
// - Zero global intermediates: only x is streamed (4 MB) and 2048 floats out.
// ---------------------------------------------------------------------------

typedef __attribute__((ext_vector_type(16))) _Float16 v16h;
typedef __attribute__((ext_vector_type(8)))  float    v8f;

#define LSTM_B     2048
#define LSTM_T     512
#define LSTM_H     50
#define LSTM_HP    64      // padded K (hidden)
#define LSTM_GP    256     // padded gates (4 x 64)
#define WG_ROWS    16      // batch rows per workgroup
#define WG_THREADS 128     // 4 waves (wave32)

// Dynamic LDS layout (bytes):
//  wp0,wi1,wp1 : 3 * 256*64 f16  = 98304
//  wih0p,bias0p,bias1p : 3*256 f32 = 3072
//  G   : 16*256 f32 = 16384
//  c0,c1,h1last : 3 * 16*64 f32 = 12288
//  xrow: 16 f32 = 64
//  h0f16,h1f16 : 2 * 16*64 f16 = 4096
#define LSTM_SMEM_BYTES (98304 + 3072 + 16384 + 12288 + 64 + 4096)

static __device__ inline float sig_f(float v) {
    return 1.0f / (1.0f + __expf(-v));
}
static __device__ inline float tanh_f(float v) {
    // tanh(v) = 2*sigmoid(2v) - 1 (fast exp path)
    return 2.0f / (1.0f + __expf(-2.0f * v)) - 1.0f;
}

static __device__ inline v8f wmma_f16(v16h a, v16h b, v8f c) {
    // (neg_a, A, neg_b, B, c_mod, C, reuse_a, reuse_b)
    return __builtin_amdgcn_wmma_f32_16x16x32_f16(false, a, false, b,
                                                  (short)0, c, false, false);
}

// Build WMMA A operand (16x32 f16, M=rows, K=hidden chunk) from a row-major
// [16][64] f16 LDS tile.  K order used here (contiguous, matches build_b):
//   lanes 0-15 : M=lane,   VGPR p in 0..7 -> K=2p,2p+1
//   lanes 16-31: same M,   K offset +16
static __device__ inline v16h build_a(const _Float16* hbuf, int lane, int kchunk) {
    union { v16h v; unsigned u[8]; } r;
    const int m  = lane & 15;
    const int kb = (lane >> 4) * 16;
    const unsigned* row = (const unsigned*)(hbuf + m * LSTM_HP + kchunk * 32);
#pragma unroll
    for (int p = 0; p < 8; ++p) {
        const int K = kb + (p << 1); // even
        r.u[p] = row[K >> 1];
    }
    return r.v;
}

// Build WMMA B operand (32x16 f16, K x N) from gate-major [256][64] f16 LDS.
// Layout: lane L holds column N = L&15; lanes 0-15 K=0..15, lanes 16-31
// K=16..31 within the chunk -> 16 contiguous K halfs = 32 contiguous bytes.
static __device__ inline v16h build_b(const _Float16* wbuf, int lane,
                                      int ntile, int kchunk) {
    union { v16h v; uint4 q[2]; } r;
    const int n  = ntile * 16 + (lane & 15);
    const int k0 = kchunk * 32 + ((lane >> 4) << 4);
    const uint4* p = (const uint4*)(wbuf + n * LSTM_HP + k0);
    r.q[0] = p[0];
    r.q[1] = p[1];
    return r.v;
}

// Store v8f accumulator (16x16 f32 C/D layout) into LDS gate buffer [16][256].
static __device__ inline void store_acc(float* g, int lane, int ntile, v8f acc) {
    const int lo = lane & 15;
    const int hi = lane >> 4;
#pragma unroll
    for (int r = 0; r < 8; ++r)
        g[(r + hi * 8) * LSTM_GP + ntile * 16 + lo] = acc[r];
}

__global__ __launch_bounds__(WG_THREADS)
void lstm2_fused_kernel(const float* __restrict__ x,
                        const float* __restrict__ Wih0, const float* __restrict__ Whh0,
                        const float* __restrict__ bih0, const float* __restrict__ bhh0,
                        const float* __restrict__ Wih1, const float* __restrict__ Whh1,
                        const float* __restrict__ bih1, const float* __restrict__ bhh1,
                        const float* __restrict__ fcW,  const float* __restrict__ fcb,
                        float* __restrict__ out) {
    extern __shared__ __align__(16) char smem[];
    _Float16* wp0    = (_Float16*)smem;                  // [256][64] W_hh0^T padded
    _Float16* wi1    = wp0 + LSTM_GP * LSTM_HP;          // [256][64] W_ih1^T padded
    _Float16* wp1    = wi1 + LSTM_GP * LSTM_HP;          // [256][64] W_hh1^T padded
    float*    wih0p  = (float*)(wp1 + LSTM_GP * LSTM_HP);// [256] layer0 input weights
    float*    bias0p = wih0p + LSTM_GP;                  // [256]
    float*    bias1p = bias0p + LSTM_GP;                 // [256]
    float*    ldsG   = bias1p + LSTM_GP;                 // [16][256] gate buffer
    float*    c0     = ldsG + WG_ROWS * LSTM_GP;         // [16][64]
    float*    c1     = c0 + WG_ROWS * LSTM_HP;           // [16][64]
    float*    h1last = c1 + WG_ROWS * LSTM_HP;           // [16][64]
    float*    xrow   = h1last + WG_ROWS * LSTM_HP;       // [16]
    _Float16* h0f16  = (_Float16*)(xrow + WG_ROWS);      // [16][64]
    _Float16* h1f16  = h0f16 + WG_ROWS * LSTM_HP;        // [16][64]

    const int tid     = threadIdx.x;
    const int lane    = tid & 31;
    const int wv      = tid >> 5;        // wave id 0..3 == gate q
    const int rowbase = blockIdx.x * WG_ROWS;

    // ---- stage + pad weights into LDS (f16), combine biases, zero state ----
    for (int idx = tid; idx < LSTM_GP * LSTM_HP; idx += WG_THREADS) {
        const int g = idx >> 6, k = idx & 63;
        const int q = g >> 6, gg = g & 63;
        const bool valid = (gg < LSTM_H) && (k < LSTM_H);
        const int src = (q * LSTM_H + gg) * LSTM_H + k;
        wp0[idx] = valid ? (_Float16)Whh0[src] : (_Float16)0.0f;
        wi1[idx] = valid ? (_Float16)Wih1[src] : (_Float16)0.0f;
        wp1[idx] = valid ? (_Float16)Whh1[src] : (_Float16)0.0f;
    }
    for (int g = tid; g < LSTM_GP; g += WG_THREADS) {
        const int q = g >> 6, gg = g & 63;
        const bool valid = gg < LSTM_H;
        const int src = q * LSTM_H + gg;
        wih0p[g]  = valid ? Wih0[src] : 0.0f;
        bias0p[g] = valid ? (bih0[src] + bhh0[src]) : 0.0f;
        bias1p[g] = valid ? (bih1[src] + bhh1[src]) : 0.0f;
    }
    for (int idx = tid; idx < WG_ROWS * LSTM_HP; idx += WG_THREADS) {
        c0[idx] = 0.0f; c1[idx] = 0.0f;
        h0f16[idx] = (_Float16)0.0f; h1f16[idx] = (_Float16)0.0f;
    }
    __syncthreads();

    // ------------------------------ time loop ------------------------------
    for (int t = 0; t < LSTM_T; ++t) {
        if (tid < WG_ROWS) {
            xrow[tid] = x[(rowbase + tid) * LSTM_T + t];
            if (t + 1 < LSTM_T)   // gfx1250 global_prefetch for next step's x
                __builtin_prefetch(&x[(rowbase + tid) * LSTM_T + t + 1], 0, 0);
        }

        // ---- layer 0: G = h0 @ W_hh0^T  (wave w computes gate q = w) ----
        {
            v16h a0 = build_a(h0f16, lane, 0);
            v16h a1 = build_a(h0f16, lane, 1);
#pragma unroll
            for (int jt = 0; jt < 4; ++jt) {
                const int n = wv * 4 + jt;
                v8f acc = {};
                acc = wmma_f16(a0, build_b(wp0, lane, n, 0), acc);
                acc = wmma_f16(a1, build_b(wp0, lane, n, 1), acc);
                store_acc(ldsG, lane, n, acc);
            }
        }
        __syncthreads();

        // ---- layer 0 elementwise: gates -> c0,h0 ----
        for (int idx = tid; idx < WG_ROWS * LSTM_HP; idx += WG_THREADS) {
            const int m = idx >> 6, j = idx & 63;
            if (j < LSTM_H) {
                const float xm = xrow[m];
                const float* gr = ldsG + m * LSTM_GP;
                const float iv = gr[j]       + xm * wih0p[j]       + bias0p[j];
                const float fv = gr[64 + j]  + xm * wih0p[64 + j]  + bias0p[64 + j];
                const float gv = gr[128 + j] + xm * wih0p[128 + j] + bias0p[128 + j];
                const float ov = gr[192 + j] + xm * wih0p[192 + j] + bias0p[192 + j];
                const float c = sig_f(fv) * c0[idx] + sig_f(iv) * tanh_f(gv);
                c0[idx] = c;
                h0f16[idx] = (_Float16)(sig_f(ov) * tanh_f(c));
            }
        }
        __syncthreads();

        // ---- layer 1: G = h0_new @ W_ih1^T + h1 @ W_hh1^T ----
        {
            v16h ax0 = build_a(h0f16, lane, 0);
            v16h ax1 = build_a(h0f16, lane, 1);
            v16h ah0 = build_a(h1f16, lane, 0);
            v16h ah1 = build_a(h1f16, lane, 1);
#pragma unroll
            for (int jt = 0; jt < 4; ++jt) {
                const int n = wv * 4 + jt;
                v8f acc = {};
                acc = wmma_f16(ax0, build_b(wi1, lane, n, 0), acc);
                acc = wmma_f16(ax1, build_b(wi1, lane, n, 1), acc);
                acc = wmma_f16(ah0, build_b(wp1, lane, n, 0), acc);
                acc = wmma_f16(ah1, build_b(wp1, lane, n, 1), acc);
                store_acc(ldsG, lane, n, acc);
            }
        }
        __syncthreads();

        // ---- layer 1 elementwise: gates -> c1,h1 ----
        for (int idx = tid; idx < WG_ROWS * LSTM_HP; idx += WG_THREADS) {
            const int m = idx >> 6, j = idx & 63;
            if (j < LSTM_H) {
                const float* gr = ldsG + m * LSTM_GP;
                const float iv = gr[j]       + bias1p[j];
                const float fv = gr[64 + j]  + bias1p[64 + j];
                const float gv = gr[128 + j] + bias1p[128 + j];
                const float ov = gr[192 + j] + bias1p[192 + j];
                const float c = sig_f(fv) * c1[idx] + sig_f(iv) * tanh_f(gv);
                c1[idx] = c;
                const float h = sig_f(ov) * tanh_f(c);
                h1f16[idx] = (_Float16)h;
                if (t == LSTM_T - 1) h1last[idx] = h;
            }
        }
        __syncthreads();
    }

    // ------------------------------ FC head --------------------------------
    if (tid < WG_ROWS) {
        float s = fcb[0];
#pragma unroll 10
        for (int j = 0; j < LSTM_H; ++j)
            s += h1last[tid * LSTM_HP + j] * fcW[j];
        out[rowbase + tid] = s;
    }
}

extern "C" void kernel_launch(void* const* d_in, const int* in_sizes, int n_in,
                              void* d_out, int out_size, void* d_ws, size_t ws_size,
                              hipStream_t stream) {
    (void)in_sizes; (void)n_in; (void)out_size; (void)d_ws; (void)ws_size;
    const float* x    = (const float*)d_in[0];
    const float* Wih0 = (const float*)d_in[1];
    const float* Whh0 = (const float*)d_in[2];
    const float* bih0 = (const float*)d_in[3];
    const float* bhh0 = (const float*)d_in[4];
    const float* Wih1 = (const float*)d_in[5];
    const float* Whh1 = (const float*)d_in[6];
    const float* bih1 = (const float*)d_in[7];
    const float* bhh1 = (const float*)d_in[8];
    const float* fcW  = (const float*)d_in[9];
    const float* fcb  = (const float*)d_in[10];
    float* out = (float*)d_out;

    dim3 grid(LSTM_B / WG_ROWS);   // 128 workgroups
    dim3 block(WG_THREADS);        // 4 waves, wave32
    lstm2_fused_kernel<<<grid, block, LSTM_SMEM_BYTES, stream>>>(
        x, Wih0, Whh0, bih0, bhh0, Wih1, Whh1, bih1, bhh1, fcW, fcb, out);
}
